// MlpMoe_60163901882987
// MI455X (gfx1250) — matrix operand order repaired and checked by reading, not compiled, not run on hardware
//
#include <hip/hip_runtime.h>

// ---------------------------------------------------------------------------
// Fused MoE SwiGLU MLP for MI455X (gfx1250, wave32, WMMA).
//   out[b] = SwiGLU(x[b] @ w1[e] + b1[e]) @ w2[e] + b2[e],  e = lead_t[b] % 8
// B=16, M=H*W=2048, C=768, HID=3072, 2*HID=6144.
//
// Memory-bound (~428 MB min HBM @ 23.3 TB/s ~ 18 us >> bf16 WMMA compute).
// Fully fused (activation never touches HBM), bf16 WMMA w/ f32 accumulation,
// coalesced weight staging into LDS pre-swizzled to the WMMA B-operand layout,
// LDS double-buffered (stage s+1 global loads overlap stage s WMMA) and
// B-operands software-pipelined through registers (ds_load t+1 overlaps wmma t).
// ---------------------------------------------------------------------------

typedef __attribute__((ext_vector_type(16))) __bf16 v16bf;
typedef __attribute__((ext_vector_type(8)))  float  v8f;

#define BATCH   16
#define MTOT    2048      // H*W rows per batch
#define CDIM    768
#define HIDD    3072
#define TWOHID  6144
#define KH      64        // hidden-dim chunk per fused step
#define MT      128       // rows per workgroup tile
#define NT      128       // output cols per workgroup tile
#define SK      64        // K-depth of one weight staging step (Phase A)
#define NSTAGE  (CDIM / SK)   // 12 stages per chunk
#define ASH_LD  (KH + 8)  // padded LDS stride (bf16 elems) for activation chunk

typedef __bf16 wbuf_t[2][8][32][16];   // one 16 KB swizzled weight block (K=64,N=128)

__device__ __forceinline__ v8f vzero8() {
    v8f z;
#pragma unroll
    for (int i = 0; i < 8; ++i) z[i] = 0.0f;
    return z;
}

// A-operand (16x32 bf16, ISA 7.12.2): lane = half*16 + m; lane holds
// K = 8*half + {0..7} in elems 0..7 and K = 16 + 8*half + {0..7} in elems 8..15.
__device__ __forceinline__ v16bf load_A_f32(const float* __restrict__ src,
                                            int ld, int lane) {
    const int half = lane >> 4, mr = lane & 15;
    const float* p = src + (size_t)mr * ld + 8 * half;
    v16bf a;
#pragma unroll
    for (int e = 0; e < 8; ++e) a[e] = (__bf16)p[e];
#pragma unroll
    for (int e = 0; e < 8; ++e) a[8 + e] = (__bf16)p[16 + e];
    return a;
}

__device__ __forceinline__ v16bf load_A_lds(const __bf16* __restrict__ src,
                                            int ld, int lane) {
    const int half = lane >> 4, mr = lane & 15;
    const __bf16* p = src + mr * ld + 8 * half;
    v16bf a;
#pragma unroll
    for (int e = 0; e < 8; ++e) a[e] = p[e];
#pragma unroll
    for (int e = 0; e < 8; ++e) a[8 + e] = p[16 + e];
    return a;
}

__device__ __forceinline__ v8f wmma_bf16(v16bf a, v16bf b, v8f c) {
    return __builtin_amdgcn_wmma_f32_16x16x32_bf16(
        /*neg_a=*/false, a, /*neg_b=*/false, b,
        /*c_mod=*/(short)0, c, /*reuse_a=*/false, /*reuse_b=*/false);
}

// Stage a K=64 x N=128 block of fc1_w (x_part cols | gate cols) into the
// WMMA-B-operand-swizzled LDS buffer. Coalesced float4 loads; bf16 pairs along
// K so the compiler merges cvt into v_cvt_pk_bf16_f32 + b32 LDS stores.
__device__ __forceinline__ void stage_w1_block(const float* __restrict__ w1,
                                               int kh, int k0, int tid,
                                               wbuf_t& buf) {
#pragma unroll
    for (int i = 0; i < 4; ++i) {
        const int u    = tid + i * 256;        // 1024 units of 2(K) x 4(N)
        const int K    = (u >> 5) * 2;         // even K in [0,64)
        const int q    = u & 31;
        const int tile = q >> 2;
        const int n4   = (q & 3) * 4;
        const int col  = (tile < 4) ? (kh + tile * 16 + n4)
                                    : (HIDD + kh + (tile - 4) * 16 + n4);
        const float* p0 = w1 + (size_t)(k0 + K) * TWOHID + col;
        const float4 a0 = *(const float4*)p0;
        const float4 a1 = *(const float4*)(p0 + TWOHID);
        const int kb   = K >> 5;
        const int lrow = ((K >> 3) & 1) * 16;
        const int ee   = (K & 7) + ((K >> 1) & 8);
        const float v0[4] = {a0.x, a0.y, a0.z, a0.w};
        const float v1[4] = {a1.x, a1.y, a1.z, a1.w};
#pragma unroll
        for (int j = 0; j < 4; ++j) {
            buf[kb][tile][lrow + n4 + j][ee]     = (__bf16)v0[j];
            buf[kb][tile][lrow + n4 + j][ee + 1] = (__bf16)v1[j];
        }
    }
}

// Stage the K=64 x N=128 fc2_w block for this chunk / col-tile.
__device__ __forceinline__ void stage_w2_block(const float* __restrict__ w2,
                                               int kh, int colbase, int tid,
                                               wbuf_t& buf) {
#pragma unroll
    for (int i = 0; i < 4; ++i) {
        const int u    = tid + i * 256;
        const int K    = (u >> 5) * 2;         // even K in [0,64)
        const int q    = u & 31;
        const int tile = q >> 2;
        const int n4   = (q & 3) * 4;
        const int col  = colbase + tile * 16 + n4;
        const float* p0 = w2 + (size_t)(kh + K) * CDIM + col;
        const float4 a0 = *(const float4*)p0;
        const float4 a1 = *(const float4*)(p0 + CDIM);
        const int kb   = K >> 5;
        const int lrow = ((K >> 3) & 1) * 16;
        const int ee   = (K & 7) + ((K >> 1) & 8);
        const float v0[4] = {a0.x, a0.y, a0.z, a0.w};
        const float v1[4] = {a1.x, a1.y, a1.z, a1.w};
#pragma unroll
        for (int j = 0; j < 4; ++j) {
            buf[kb][tile][lrow + n4 + j][ee]     = (__bf16)v0[j];
            buf[kb][tile][lrow + n4 + j][ee + 1] = (__bf16)v1[j];
        }
    }
}

__global__ __launch_bounds__(256, 1)
void moe_mlp_fused(const float* __restrict__ x,
                   const int*   __restrict__ lead_t,
                   const float* __restrict__ fc1_w,
                   const float* __restrict__ fc1_b,
                   const float* __restrict__ fc2_w,
                   const float* __restrict__ fc2_b,
                   float* __restrict__ out) {
    const int b  = blockIdx.z;          // batch element
    const int mt = blockIdx.y;          // row tile (128 rows)
    const int nt = blockIdx.x;          // output col tile (128 cols)
    const int e  = lead_t[b] & 7;       // expert route

    const int tid  = threadIdx.x;
    const int wave = tid >> 5;          // 8 waves
    const int lane = tid & 31;
    const int row0 = wave * 16;         // wave's rows within the 128-row tile

    const float* xb = x + ((size_t)b * MTOT + (size_t)mt * MT) * CDIM;
    const float* w1 = fc1_w + (size_t)e * CDIM * TWOHID;
    const float* b1 = fc1_b + (size_t)e * TWOHID;
    const float* w2 = fc2_w + (size_t)e * HIDD * CDIM;
    const float* b2 = fc2_b + (size_t)e * CDIM;
    float* outb = out + ((size_t)b * MTOT + (size_t)mt * MT) * CDIM + nt * NT;

    // Ping-pong swizzled weight buffers:
    //   wB[p][kb][tile][lane][e] <-> B element (K = 32*kb + 8*(lane/16) + e%8 + 16*(e/8),
    //                                            N = 16*tile + lane%16)
    __shared__ __align__(32) __bf16 wB[2][2][8][32][16];   // 2 x 16 KB
    __shared__ __align__(16) __bf16 a_sh[MT][ASH_LD];      // 18 KB activations

    v8f oacc[8];
#pragma unroll
    for (int t = 0; t < 8; ++t) oacc[t] = vzero8();

    const int hlf = lane >> 4;
    const int nn  = lane & 15;

    for (int kh = 0; kh < HIDD; kh += KH) {
        // =========== Phase A: h = x_tile @ w1[:, chunk] (x_part & gate) ===========
        v8f hx[4], hg[4];
#pragma unroll
        for (int t = 0; t < 4; ++t) { hx[t] = vzero8(); hg[t] = vzero8(); }

        // Pull this chunk's w2 block toward L2 while Phase A computes.
        __builtin_prefetch(w2 + (size_t)(kh + (tid >> 2)) * CDIM
                              + nt * NT + (tid & 3) * 32, 0, 3);

        __syncthreads();   // wB[0] free of previous chunk's Phase B readers
        stage_w1_block(w1, kh, 0, tid, wB[0]);       // prologue: stage s=0

        for (int s = 0; s < NSTAGE; ++s) {           // 12 stages of K=64
            const int k0 = s * SK;
            __syncthreads();                         // buffer s&1 is staged
            if (s + 1 < NSTAGE)                      // overlap: loads for s+1
                stage_w1_block(w1, kh, k0 + SK, tid, wB[(s + 1) & 1]);
#pragma unroll
            for (int ks = 0; ks < SK / 32; ++ks) {
                v16bf a = load_A_f32(xb + (size_t)row0 * CDIM + k0 + 32 * ks,
                                     CDIM, lane);
                // Software-pipelined B operands: ds_load of tile t+1 is in
                // flight while wmma of tile t executes (dscnt partial waits).
                v16bf bnext = *(const v16bf*)&wB[s & 1][ks][0][lane][0];
#pragma unroll
                for (int t = 0; t < 8; ++t) {
                    v16bf bcur = bnext;
                    if (t < 7)
                        bnext = *(const v16bf*)&wB[s & 1][ks][t + 1][lane][0];
                    if (t < 4) hx[t] = wmma_bf16(a, bcur, hx[t]);
                    else       hg[t - 4] = wmma_bf16(a, bcur, hg[t - 4]);
                }
            }
        }

        __syncthreads();   // Phase A readers of wB done; prev a_sh readers done

        // ---------- SwiGLU (+fc1 bias) in registers, spill chunk to LDS ----------
#pragma unroll
        for (int t = 0; t < 4; ++t) {
            const int col = kh + 16 * t + nn;          // hidden index (x_part col)
            const float bx1 = b1[col];
            const float bg1 = b1[HIDD + col];
#pragma unroll
            for (int r = 0; r < 8; ++r) {
                const float xp = hx[t][r] + bx1;
                const float g  = hg[t][r] + bg1;
                const float si = g / (1.0f + __expf(-g));   // silu(gate)
                const int row  = row0 + r + 8 * hlf;        // C/D layout row
                a_sh[row][16 * t + nn] = (__bf16)(si * xp);
            }
        }

        // Stage w2[kh:kh+64, nt*128:+128] into wB[0] within the same window.
        stage_w2_block(w2, kh, nt * NT, tid, wB[0]);
        __syncthreads();

        // ---------- Phase B: oacc += a_chunk @ w2[chunk, cols] ----------
#pragma unroll
        for (int ks = 0; ks < KH / 32; ++ks) {
            v16bf a = load_A_lds(&a_sh[row0][32 * ks], ASH_LD, lane);
            v16bf bnext = *(const v16bf*)&wB[0][ks][0][lane][0];
#pragma unroll
            for (int t = 0; t < 8; ++t) {
                v16bf bcur = bnext;
                if (t < 7)
                    bnext = *(const v16bf*)&wB[0][ks][t + 1][lane][0];
                oacc[t] = wmma_bf16(a, bcur, oacc[t]);
            }
        }
    }

    // ---------- Epilogue: +fc2 bias, store fp32 ----------
#pragma unroll
    for (int t = 0; t < 8; ++t) {
        const float bias = b2[nt * NT + 16 * t + nn];
#pragma unroll
        for (int r = 0; r < 8; ++r) {
            const int row = row0 + r + 8 * hlf;
            outb[(size_t)row * CDIM + 16 * t + nn] = oacc[t][r] + bias;
        }
    }
}

extern "C" void kernel_launch(void* const* d_in, const int* in_sizes, int n_in,
                              void* d_out, int out_size, void* d_ws, size_t ws_size,
                              hipStream_t stream) {
    (void)in_sizes; (void)n_in; (void)d_ws; (void)ws_size; (void)out_size;
    const float* x      = (const float*)d_in[0];
    const int*   lead_t = (const int*)d_in[1];
    const float* fc1_w  = (const float*)d_in[2];
    const float* fc1_b  = (const float*)d_in[3];
    const float* fc2_w  = (const float*)d_in[4];
    const float* fc2_b  = (const float*)d_in[5];
    float* out = (float*)d_out;

    dim3 grid(CDIM / NT, MTOT / MT, BATCH);   // (6, 16, 16)
    dim3 block(256);
    moe_mlp_fused<<<grid, block, 0, stream>>>(x, lead_t, fc1_w, fc1_b,
                                              fc2_w, fc2_b, out);
}
